// MLAttention_3624952398463
// MI455X (gfx1250) — compile-verified, matrix operand order checked by reading, not run on hardware
//
#include <hip/hip_runtime.h>

typedef __attribute__((ext_vector_type(16))) __bf16        v16bf;
typedef __attribute__((ext_vector_type(8)))  float         v8f;
typedef __attribute__((ext_vector_type(4)))  unsigned int  u32x4;

union FragBF { v16bf v; u32x4 q[2]; };

__device__ __forceinline__ unsigned short f32_to_bf16(float f) {
  unsigned int u = __float_as_uint(f);
  u += 0x7FFFu + ((u >> 16) & 1u);   // round-to-nearest-even
  return (unsigned short)(u >> 16);
}

// wave32: xor-swizzle within 16-lane groups (masks 1,2,4,8 never cross halves)
#define SWZ_XOR(x, m) \
  __int_as_float(__builtin_amdgcn_ds_swizzle(__float_as_int(x), (0x1F | ((m) << 10))))

// CDNA5 async global->LDS copy (ASYNCcnt-tracked, no VGPR data round-trip).
// lds32 = low 32 bits of generic LDS pointer (= LDS byte address, ISA 10.2).
#define ASYNC_G2L_B128(lds32, gptr)                                   \
  asm volatile("global_load_async_to_lds_b128 %0, %1, off"            \
               :: "v"(lds32), "v"(gptr) : "memory")
#define WAIT_ASYNC0() asm volatile("s_wait_asynccnt 0x0" ::: "memory")

__device__ __forceinline__ unsigned lds_addr(const void* p) {
  return (unsigned)(unsigned long long)p;
}

// ---------------------------------------------------------------------------
// fp32 -> bf16 convert (n multiple of 1024)
// ---------------------------------------------------------------------------
__global__ __launch_bounds__(256) void cvt_f32_to_bf16(
    const float* __restrict__ in, unsigned short* __restrict__ out, int n) {
  const int i = (blockIdx.x * blockDim.x + threadIdx.x) * 4;
  if (i + 3 < n) {
#pragma unroll
    for (int j = 0; j < 4; ++j) out[i + j] = f32_to_bf16(in[i + j]);
  }
}

// ---------------------------------------------------------------------------
// C[MxN] = A[MxK] * B[KxN], bf16 in, bf16 and/or f32 out.
// 128x128 block tile, 8 waves (4x2), wave tile 32x64, K-step 32.
// A tile staged via async global->LDS; B tile transposed with packed-pair b32.
// ---------------------------------------------------------------------------
__global__ __launch_bounds__(256) void gemm_bf16_128x128(
    const unsigned short* __restrict__ A,
    const unsigned short* __restrict__ Bm,
    unsigned short* __restrict__ Cb,   // bf16 out (may be null)
    float* __restrict__ Cf,            // f32 out (may be null)
    int M, int N, int K) {
  __shared__ unsigned short As[128 * 32];  // [m][k]
  __shared__ unsigned short Bs[128 * 32];  // transposed: [n][k]

  const int tid  = threadIdx.x;
  const int wave = tid >> 5, lane = tid & 31;
  const int l16  = lane & 15, lh = lane >> 4;
  const int wm   = wave >> 1, wn = wave & 1;
  const int rowBase = blockIdx.y * 128;
  const int colBase = blockIdx.x * 128;

  v8f acc[2][4];
#pragma unroll
  for (int i = 0; i < 2; ++i)
#pragma unroll
    for (int j = 0; j < 4; ++j) acc[i][j] = {};

  const int ar  = tid >> 1;          // A row per thread (two 16B chunks)
  const int ac  = (tid & 1) * 16;
  const int bkp = (tid >> 4) * 2;    // B k-row pair: 2p, 2p+1
  const int bn  = (tid & 15) * 8;    // 8 columns per thread

  for (int k0 = 0; k0 < K; k0 += 32) {
    {  // A tile 128x32: async copy straight into LDS
      const unsigned short* src = A + (size_t)(rowBase + ar) * K + k0 + ac;
      const unsigned la = lds_addr(As + ar * 32 + ac);
      ASYNC_G2L_B128(la, src);
      ASYNC_G2L_B128(la + 16u, src + 8);
    }
    {  // B tile 32x128 -> Bs[n][k], packing k-pairs into b32 stores
      const unsigned short* s0 = Bm + (size_t)(k0 + bkp) * N + colBase + bn;
      u32x4 r0 = *(const u32x4*)(s0);
      u32x4 r1 = *(const u32x4*)(s0 + N);
      unsigned short a0[8], a1[8];
      *(u32x4*)a0 = r0;
      *(u32x4*)a1 = r1;
#pragma unroll
      for (int j = 0; j < 8; ++j) {
        const unsigned packed = (unsigned)a0[j] | ((unsigned)a1[j] << 16);
        *(unsigned*)(Bs + (bn + j) * 32 + bkp) = packed;
      }
    }
    if (k0 + 32 < K) {  // global_prefetch_b8 of next tiles (L2-resident set)
      __builtin_prefetch(A + (size_t)(rowBase + ar) * K + k0 + 32 + ac, 0, 2);
      __builtin_prefetch(Bm + (size_t)(k0 + 32 + bkp) * N + colBase + bn, 0, 2);
    }
    WAIT_ASYNC0();
    __syncthreads();

    FragBF af[2], bf[4];
#pragma unroll
    for (int i = 0; i < 2; ++i) {
      const int r = wm * 32 + i * 16 + l16;
      af[i].q[0] = *(const u32x4*)(As + r * 32 + lh * 8);
      af[i].q[1] = *(const u32x4*)(As + r * 32 + 16 + lh * 8);
    }
#pragma unroll
    for (int j = 0; j < 4; ++j) {
      const int n = wn * 64 + j * 16 + l16;
      bf[j].q[0] = *(const u32x4*)(Bs + n * 32 + lh * 8);
      bf[j].q[1] = *(const u32x4*)(Bs + n * 32 + 16 + lh * 8);
    }
#pragma unroll
    for (int i = 0; i < 2; ++i)
#pragma unroll
      for (int j = 0; j < 4; ++j)
        acc[i][j] = __builtin_amdgcn_wmma_f32_16x16x32_bf16(
            false, af[i].v, false, bf[j].v, (short)0, acc[i][j], false, false);
    __syncthreads();
  }

#pragma unroll
  for (int i = 0; i < 2; ++i)
#pragma unroll
    for (int j = 0; j < 4; ++j)
#pragma unroll
      for (int e = 0; e < 8; ++e) {
        const int row = rowBase + wm * 32 + i * 16 + e + lh * 8;
        const int col = colBase + wn * 64 + j * 16 + l16;
        const float vv = acc[i][j][e];
        if (Cb) Cb[(size_t)row * N + col] = f32_to_bf16(vv);
        if (Cf) Cf[(size_t)row * N + col] = vv;
      }
}

// ---------------------------------------------------------------------------
// Causal flash attention, hd=128, 128-query block per workgroup,
// 8 waves x 16 query rows. K/V^T share one 32KB LDS buffer (phased).
// K tile staged via async global->LDS; V^T via packed-pair b32 transpose.
// ---------------------------------------------------------------------------
__global__ __launch_bounds__(256) void mla_attn_128(
    const unsigned short* __restrict__ Qm,  // [B*T, C] bf16
    const unsigned short* __restrict__ Km,
    const unsigned short* __restrict__ Vm,
    unsigned short* __restrict__ Om,        // [B*T, C] bf16
    int T, int C) {
  __shared__ unsigned short KVs[128 * 128];    // phase1: K [key][d]; phase2: V^T [d][key]
  __shared__ unsigned short Ps[8 * 16 * 128];  // per-wave P (16x128) bf16

  const int qb = blockIdx.x, h = blockIdx.y, b = blockIdx.z;
  const int tid  = threadIdx.x;
  const int wave = tid >> 5, lane = tid & 31;
  const int l16  = lane & 15, lh = lane >> 4;

  const size_t headOff = (size_t)h * 128;
  const size_t baseRow = (size_t)b * T;

  // preload this wave's Q fragments (16 rows x hd=128) from global
  FragBF qf[4];
  {
    const unsigned short* qrow =
        Qm + (baseRow + qb * 128 + wave * 16 + l16) * (size_t)C + headOff;
#pragma unroll
    for (int ks = 0; ks < 4; ++ks) {
      qf[ks].q[0] = *(const u32x4*)(qrow + ks * 32 + lh * 8);
      qf[ks].q[1] = *(const u32x4*)(qrow + ks * 32 + 16 + lh * 8);
    }
  }

  float m_run[8], l_run[8];
  v8f   o_acc[8];
#pragma unroll
  for (int e = 0; e < 8; ++e) { m_run[e] = -3.0e38f; l_run[e] = 0.f; }
#pragma unroll
  for (int t = 0; t < 8; ++t) o_acc[t] = {};

  const float scale = 0.08838834764831845f;  // 1/sqrt(128)
  const int   qrow0 = qb * 128 + wave * 16;

  for (int kb = 0; kb <= qb; ++kb) {
    __syncthreads();  // all waves done with previous V^T
    {                 // load K tile [key][d] via async global->LDS
      const unsigned short* kbase = Km + (baseRow + kb * 128) * (size_t)C + headOff;
      for (int c = tid; c < 2048; c += 256) {
        const int r = c >> 4, kk = (c & 15) * 8;
        ASYNC_G2L_B128(lds_addr(KVs + r * 128 + kk), kbase + (size_t)r * C + kk);
      }
    }
    WAIT_ASYNC0();
    __syncthreads();

    // S = Q K^T  (16 x 128 per wave)
    v8f s[8];
#pragma unroll
    for (int nt = 0; nt < 8; ++nt) s[nt] = {};
#pragma unroll
    for (int ks = 0; ks < 4; ++ks) {
      const int kk = ks * 32;
#pragma unroll
      for (int nt = 0; nt < 8; ++nt) {
        FragBF kf;
        const int n = nt * 16 + l16;
        kf.q[0] = *(const u32x4*)(KVs + n * 128 + kk + lh * 8);
        kf.q[1] = *(const u32x4*)(KVs + n * 128 + kk + 16 + lh * 8);
        s[nt] = __builtin_amdgcn_wmma_f32_16x16x32_bf16(
            false, qf[ks].v, false, kf.v, (short)0, s[nt], false, false);
      }
    }
    __syncthreads();  // done reading K -> buffer reusable

    {  // load V^T tile [d][key]: packed key-pair b32 transpose
      const unsigned short* vbase = Vm + (baseRow + kb * 128) * (size_t)C + headOff;
      for (int c = tid; c < 1024; c += 256) {
        const int key = (c >> 4) * 2;      // key pair 2p, 2p+1
        const int dd  = (c & 15) * 8;      // 8 d values
        const unsigned short* s0 = vbase + (size_t)key * C + dd;
        u32x4 r0 = *(const u32x4*)(s0);
        u32x4 r1 = *(const u32x4*)(s0 + C);
        unsigned short a0[8], a1[8];
        *(u32x4*)a0 = r0;
        *(u32x4*)a1 = r1;
#pragma unroll
        for (int j = 0; j < 8; ++j) {
          const unsigned packed = (unsigned)a0[j] | ((unsigned)a1[j] << 16);
          *(unsigned*)(KVs + (dd + j) * 128 + key) = packed;
        }
      }
    }

    // online softmax (rows split: lanes 0-15 -> rows e, lanes 16-31 -> rows e+8)
    float mnew[8];
#pragma unroll
    for (int e = 0; e < 8; ++e) {
      const int qg = qrow0 + e + lh * 8;
#pragma unroll
      for (int nt = 0; nt < 8; ++nt) {
        float vv = s[nt][e] * scale;
        const int kg = kb * 128 + nt * 16 + l16;
        if (kg > qg) vv = -3.0e38f;  // causal mask
        s[nt][e] = vv;
      }
      float mx = s[0][e];
#pragma unroll
      for (int nt = 1; nt < 8; ++nt) mx = fmaxf(mx, s[nt][e]);
      mx = fmaxf(mx, SWZ_XOR(mx, 1));
      mx = fmaxf(mx, SWZ_XOR(mx, 2));
      mx = fmaxf(mx, SWZ_XOR(mx, 4));
      mx = fmaxf(mx, SWZ_XOR(mx, 8));
      mnew[e] = fmaxf(m_run[e], mx);
    }
    float rsum[8];
#pragma unroll
    for (int e = 0; e < 8; ++e) rsum[e] = 0.f;
#pragma unroll
    for (int nt = 0; nt < 8; ++nt)
#pragma unroll
      for (int e = 0; e < 8; ++e) {
        const float p = __expf(s[nt][e] - mnew[e]);
        rsum[e] += p;
        Ps[wave * 2048 + (e + lh * 8) * 128 + nt * 16 + l16] = f32_to_bf16(p);
      }
#pragma unroll
    for (int e = 0; e < 8; ++e) {
      float rs = rsum[e];
      rs += SWZ_XOR(rs, 1);
      rs += SWZ_XOR(rs, 2);
      rs += SWZ_XOR(rs, 4);
      rs += SWZ_XOR(rs, 8);
      const float corr = __expf(m_run[e] - mnew[e]);
      l_run[e] = l_run[e] * corr + rs;
      m_run[e] = mnew[e];
#pragma unroll
      for (int ot = 0; ot < 8; ++ot) o_acc[ot][e] *= corr;
    }
    __syncthreads();  // V^T visible to all (Ps is same-wave: LDS in-order)

    // O += P V
#pragma unroll
    for (int ks = 0; ks < 4; ++ks) {
      const int kk = ks * 32;
      FragBF pf;
      pf.q[0] = *(const u32x4*)(Ps + wave * 2048 + l16 * 128 + kk + lh * 8);
      pf.q[1] = *(const u32x4*)(Ps + wave * 2048 + l16 * 128 + kk + 16 + lh * 8);
#pragma unroll
      for (int ot = 0; ot < 8; ++ot) {
        FragBF vf;
        const int d = ot * 16 + l16;
        vf.q[0] = *(const u32x4*)(KVs + d * 128 + kk + lh * 8);
        vf.q[1] = *(const u32x4*)(KVs + d * 128 + kk + 16 + lh * 8);
        o_acc[ot] = __builtin_amdgcn_wmma_f32_16x16x32_bf16(
            false, pf.v, false, vf.v, (short)0, o_acc[ot], false, false);
      }
    }
  }

  // normalized output -> [B*T, C] bf16
  unsigned short* obase = Om + (baseRow + qb * 128 + wave * 16) * (size_t)C + headOff;
#pragma unroll
  for (int ot = 0; ot < 8; ++ot)
#pragma unroll
    for (int e = 0; e < 8; ++e) {
      const int row  = e + lh * 8;
      const int dcol = ot * 16 + l16;
      obase[(size_t)row * C + dcol] = f32_to_bf16(o_acc[ot][e] / l_run[e]);
    }
}

// ---------------------------------------------------------------------------
extern "C" void kernel_launch(void* const* d_in, const int* in_sizes, int n_in,
                              void* d_out, int out_size, void* d_ws, size_t ws_size,
                              hipStream_t stream) {
  constexpr int B = 2, T = 2048, D = 2048, L = 512, H = 16;
  constexpr int BT = B * T;
  (void)in_sizes; (void)n_in; (void)out_size; (void)ws_size;

  const float* x     = (const float*)d_in[0];
  const float* Wq    = (const float*)d_in[1];
  const float* Wdown = (const float*)d_in[2];
  const float* Wkup  = (const float*)d_in[3];
  const float* Wvup  = (const float*)d_in[4];
  const float* Wout  = (const float*)d_in[5];

  float* out_proj = (float*)d_out;                 // [B,T,D]
  float* out_lat  = out_proj + (size_t)BT * D;     // [B,T,L]

  char*  ws  = (char*)d_ws;
  size_t off = 0;
  auto alloc = [&](size_t elems) {
    unsigned short* p = (unsigned short*)(ws + off);
    off += ((elems * 2 + 255) & ~((size_t)255));
    return p;
  };
  unsigned short* xb    = alloc((size_t)BT * D);
  unsigned short* Wqb   = alloc((size_t)D * D);
  unsigned short* Wdb   = alloc((size_t)D * L);
  unsigned short* Wkb   = alloc((size_t)L * D);
  unsigned short* Wvb   = alloc((size_t)L * D);
  unsigned short* Wob   = alloc((size_t)D * D);
  unsigned short* qbuf  = alloc((size_t)BT * D);
  unsigned short* latb  = alloc((size_t)BT * L);
  unsigned short* kbuf  = alloc((size_t)BT * D);
  unsigned short* vbuf  = alloc((size_t)BT * D);
  unsigned short* aobuf = alloc((size_t)BT * D);

  auto cvt = [&](const float* src, unsigned short* dst, size_t n) {
    cvt_f32_to_bf16<<<(unsigned)(n / 1024), 256, 0, stream>>>(src, dst, (int)n);
  };
  cvt(x,     xb,  (size_t)BT * D);
  cvt(Wq,    Wqb, (size_t)D * D);
  cvt(Wdown, Wdb, (size_t)D * L);
  cvt(Wkup,  Wkb, (size_t)L * D);
  cvt(Wvup,  Wvb, (size_t)L * D);
  cvt(Wout,  Wob, (size_t)D * D);

  dim3 blk(256);
  // q = x @ Wq
  gemm_bf16_128x128<<<dim3(D / 128, BT / 128), blk, 0, stream>>>(xb, Wqb, qbuf, nullptr, BT, D, D);
  // latent = x @ Wkv_down  (bf16 for k/v GEMMs + fp32 straight to output)
  gemm_bf16_128x128<<<dim3(L / 128, BT / 128), blk, 0, stream>>>(xb, Wdb, latb, out_lat, BT, L, D);
  // k = latent @ Wk_up ; v = latent @ Wv_up
  gemm_bf16_128x128<<<dim3(D / 128, BT / 128), blk, 0, stream>>>(latb, Wkb, kbuf, nullptr, BT, D, L);
  gemm_bf16_128x128<<<dim3(D / 128, BT / 128), blk, 0, stream>>>(latb, Wvb, vbuf, nullptr, BT, D, L);
  // causal attention
  mla_attn_128<<<dim3(T / 128, H, B), blk, 0, stream>>>(qbuf, kbuf, vbuf, aobuf, T, D);
  // out = attn @ Wout (fp32 to d_out)
  gemm_bf16_128x128<<<dim3(D / 128, BT / 128), blk, 0, stream>>>(aobuf, Wob, nullptr, out_proj, BT, D, D);
}
